// SceneSelector_21509196218821
// MI455X (gfx1250) — matrix operand order repaired and checked by reading, not compiled, not run on hardware
//
#include <hip/hip_runtime.h>
#include <math.h>
#include <stdint.h>

typedef __attribute__((ext_vector_type(16))) _Float16 v16h;
typedef __attribute__((ext_vector_type(8)))  _Float16 v8h;
typedef __attribute__((ext_vector_type(8)))  float    v8f;

#define IMG      256
#define TILE     64
#define NTHREADS 256
#define NWAVES   8

// padded LDS plane geometry: [3 channels][84 rows][88 cols] f16
#define PROWS 84
#define PCOLS 88
#define PLANE (PROWS * PCOLS)      // 7392
#define PSZ   (3 * PLANE)          // 22176 halves per activation buffer

#define R_IN  76                   // valid input region (64 + 2*6 halo)
#define R_H1  72
#define R_H2  68

// ---------------------------------------------------------------------------
// CDNA5 async global->LDS copy + dedicated counter wait (ISA 08_async_tensor).
// GVS mode: SGPR 64-bit base + per-lane VGPR byte offset; VDST VGPR carries
// the LDS byte address (low 32 bits of the generic LDS pointer).
// ---------------------------------------------------------------------------
__device__ __forceinline__ void async_ld_f32(float* lds_dst, const float* g_base, int gelem) {
  unsigned lds_addr = (unsigned)(uintptr_t)lds_dst;
  asm volatile("global_load_async_to_lds_b32 %0, %1, %2"
               :: "v"(lds_addr), "v"(gelem * 4), "s"(g_base)
               : "memory");
}
__device__ __forceinline__ void wait_asynccnt0() {
  asm volatile("s_wait_asynccnt 0x0" ::: "memory");
}

// ---------------------------------------------------------------------------
// Banded-GEMM K mapping: k = (c,dy)*24 + t, t = absolute column offset 0..23.
// LDS half-offset for an 8-aligned run starting at k (row/tile base added later).
// ---------------------------------------------------------------------------
__device__ constexpr int Fk(int k) {
  return ((k / 24) / 5) * PLANE + ((k / 24) % 5) * PCOLS + (k % 24);
}
__device__ constexpr int FkSafe(int k) { return k < 360 ? Fk(k) : 0; }

// ---------------------------------------------------------------------------
// Stage conv weights (OC x 3 x 5 x 5 fp32) into banded B-operand layout:
// B is 32x16 (K x N); lane n = lane&15, element i -> K = (lane&16?8:0)+(i<8?i:i+8).
// wB[oc][12 chunks][32 lanes][16 halves];  B[k,n] = w[oc][c][dy][t-n] in-band.
// ---------------------------------------------------------------------------
__device__ __forceinline__ void stage_wB_banded(_Float16* wB, const float* wg,
                                                int OC, int tid) {
  const int total = OC * 12 * 32 * 16;
  for (int idx = tid; idx < total; idx += NTHREADS) {
    int i      = idx & 15;
    int lane_b = (idx >> 4) & 31;
    int chunk  = (idx >> 9) % 12;
    int oc     = idx / 6144;
    int n  = lane_b & 15;
    int kl = ((lane_b & 16) ? 8 : 0) + (i < 8 ? i : i + 8);
    int k  = chunk * 32 + kl;
    float v = 0.0f;
    if (k < 360) {
      int p = k / 24, t = k % 24;
      int cin = p / 5, dy = p % 5;
      int dx  = t - n;
      if (dx >= 0 && dx <= 4)
        v = wg[((oc * 3 + cin) * 5 + dy) * 5 + dx];
    }
    wB[idx] = (_Float16)v;
  }
}

// ---------------------------------------------------------------------------
// One conv layer, banded implicit GEMM on 16x16 output tiles:
// 12 WMMA K-steps per (tile, oc); A-operand = two aligned ds_load_b128/step.
// ReLU + bias; out-of-image positions forced to zero (SAME padding of the
// next layer). dstH != nullptr -> padded f16 plane; else f32 64x64 tile.
// ---------------------------------------------------------------------------
__device__ __forceinline__ void conv_banded(const _Float16* __restrict__ src,
                                            _Float16* __restrict__ dstH,
                                            float* __restrict__ dstF,
                                            int OC, int Rout, int ntx, int ntiles,
                                            const _Float16* __restrict__ wB,
                                            const float* __restrict__ sbias,
                                            const int* offR0, const int* offR1, int hb,
                                            int wave, int lane,
                                            int gy0, int gx0, bool mask) {
  const int m  = lane & 15;                 // A row / D column index
  const int mb = (lane & 16) ? 8 : 0;       // D row offset for this lane half
  const int ntasks = ntiles * OC;
  const v8h z = {};

  for (int task = wave; task < ntasks; task += NWAVES) {
    int oc  = task / ntiles;
    int r   = task % ntiles;
    int py0 = (r / ntx) * 16;
    int px0 = (r % ntx) * 16;
    int rowbase = (py0 + m) * PCOLS + px0;
    const _Float16* wb = wB + oc * 6144 + lane * 16;

    v8f acc = {};
#pragma unroll
    for (int c = 0; c < 11; ++c) {
      union { v16h v; v8h h[2]; } a;
      a.h[0] = *(const v8h*)(src + offR0[c] + rowbase);   // ds_load_b128
      a.h[1] = *(const v8h*)(src + offR1[c] + rowbase);   // ds_load_b128
      v16h b = *(const v16h*)(wb + c * 512);
      acc = __builtin_amdgcn_wmma_f32_16x16x32_f16(false, a.v, false, b,
                                                   (short)0, acc, false, false);
    }
    { // chunk 11: only k=352..359 live (lane-half hb==0); rest is K padding
      union { v16h v; v8h h[2]; } a;
      v8h t0 = *(const v8h*)(src + offR0[11] + rowbase);
      a.h[0] = hb ? z : t0;
      a.h[1] = z;
      v16h b = *(const v16h*)(wb + 11 * 512);
      acc = __builtin_amdgcn_wmma_f32_16x16x32_f16(false, a.v, false, b,
                                                   (short)0, acc, false, false);
    }

    float bias = sbias[oc];
#pragma unroll
    for (int j = 0; j < 8; ++j) {
      int row = py0 + mb + j;
      int col = px0 + m;
      if (row < Rout && col < Rout) {
        float v = acc[j] + bias;
        v = v > 0.0f ? v : 0.0f;                          // ReLU
        if (mask) {
          int gy = gy0 + row, gx = gx0 + col;
          if ((unsigned)gy >= IMG || (unsigned)gx >= IMG) v = 0.0f;
        }
        if (dstH) dstH[(oc * PROWS + row) * PCOLS + col] = (_Float16)v;
        else      dstF[row * TILE + col] = v;
      }
    }
  }
}

// ---------------------------------------------------------------------------
// Kernel A: fused conv1+conv2+conv3 + 16x16 patch-mean pooling.
// One block per (image, 64x64 tile). x arrives via async global->LDS DMA;
// all intermediates stay in LDS. LDS pool is reused phase-by-phase:
//   phase 0: fp32 x staging (69312 B)
//   later  : wB (36864 B) | h3 f32 (16384 B) | psum (1024 B)
// ---------------------------------------------------------------------------
__global__ __launch_bounds__(NTHREADS)
void conv_pool_fused_kernel(const float* __restrict__ x,
                            const float* __restrict__ w1, const float* __restrict__ b1,
                            const float* __restrict__ w2, const float* __restrict__ b2,
                            const float* __restrict__ w3, const float* __restrict__ b3,
                            float* __restrict__ feat) {
  __shared__ __align__(32) char     s_pool[3 * R_IN * R_IN * 4];  // 69312 B
  __shared__ __align__(32) _Float16 s_x [PSZ];                    // 44352 B
  __shared__ __align__(32) _Float16 s_h1[PSZ];                    // 44352 B
  __shared__ __align__(32) _Float16 s_h2[PSZ];                    // 44352 B
  __shared__ float s_bias[4];

  float*    s_stagef = (float*)s_pool;               // phase 0
  _Float16* s_wB     = (_Float16*)s_pool;            // phases 2+ (<= 36864 B)
  float*    s_h3     = (float*)(s_pool + 36864);     // 16384 B
  float*    s_psum   = (float*)(s_pool + 53248);     // 1024 B

  const int tile = blockIdx.x;          // 0..15
  const int img  = blockIdx.y;          // 0..255
  const int tx   = (tile & 3) * TILE;
  const int ty   = (tile >> 2) * TILE;
  const int tid  = threadIdx.x;
  const int wave = tid >> 5;
  const int lane = tid & 31;
  const int hb   = (lane >> 4) & 1;

  // A-run LDS offsets: depend only on (chunk, lane-half); constexpr-folded.
  int offR0[12], offR1[12];
#pragma unroll
  for (int c = 0; c < 12; ++c) {
    offR0[c] = hb ? FkSafe(32 * c + 8)  : FkSafe(32 * c);
    offR1[c] = hb ? FkSafe(32 * c + 24) : FkSafe(32 * c + 16);
  }

  // ---- phase 0: async-DMA the haloed input tile (fp32) into the LDS pool,
  //      zero the padded margins of h1/h2 (written once, read by next layer) --
  const float* xg = x + (size_t)img * (3 * IMG * IMG);
  for (int idx = tid; idx < 3 * R_IN * R_IN; idx += NTHREADS) {
    int c  = idx / (R_IN * R_IN);
    int r  = idx % (R_IN * R_IN);
    int ly = r / R_IN, lx = r % R_IN;
    int gy = ty - 6 + ly, gx = tx - 6 + lx;
    if ((unsigned)gy < IMG && (unsigned)gx < IMG)
      async_ld_f32(s_stagef + idx, xg, (c * IMG + gy) * IMG + gx);
    else
      s_stagef[idx] = 0.0f;
  }
  {
    uint4 zz = {0, 0, 0, 0};
    uint4* z1 = (uint4*)s_h1;
    uint4* z2 = (uint4*)s_h2;
    for (int i = tid; i < (int)(PSZ * 2 / 16); i += NTHREADS) { z1[i] = zz; z2[i] = zz; }
  }
  if (tid == 0 && img + 1 < IMG)                     // warm L2 for the next image
    __builtin_prefetch(xg + 3 * IMG * IMG, 0, 0);    // -> global_prefetch_b8

  wait_asynccnt0();
  __syncthreads();

  // ---- phase 1: convert staged fp32 -> padded f16 plane ----
  for (int idx = tid; idx < PSZ; idx += NTHREADS) {
    int cc  = idx / PLANE;
    int r2  = idx % PLANE;
    int row = r2 / PCOLS, col = r2 % PCOLS;
    float v = 0.0f;
    if (row < R_IN && col < R_IN)
      v = s_stagef[(cc * R_IN + row) * R_IN + col];
    s_x[idx] = (_Float16)v;
  }
  __syncthreads();                                   // staging pool now free

  // ---- conv1: x -> h1 (72x72 valid), 5x5 tiles, 3 oc ----
  stage_wB_banded(s_wB, w1, 3, tid);
  if (tid < 3) s_bias[tid] = b1[tid];
  __syncthreads();
  conv_banded(s_x, s_h1, nullptr, 3, R_H1, 5, 25, s_wB, s_bias,
              offR0, offR1, hb, wave, lane, ty - 4, tx - 4, true);
  __syncthreads();

  // ---- conv2: h1 -> h2 (68x68 valid) ----
  stage_wB_banded(s_wB, w2, 3, tid);
  if (tid < 3) s_bias[tid] = b2[tid];
  __syncthreads();
  conv_banded(s_h1, s_h2, nullptr, 3, R_H2, 5, 25, s_wB, s_bias,
              offR0, offR1, hb, wave, lane, ty - 2, tx - 2, true);
  __syncthreads();

  // ---- conv3: h2 -> h3 (64x64 f32), 4x4 tiles, 1 oc ----
  stage_wB_banded(s_wB, w3, 1, tid);
  if (tid == 0) s_bias[0] = b3[0];
  __syncthreads();
  conv_banded(s_h2, nullptr, s_h3, 1, TILE, 4, 16, s_wB, s_bias,
              offR0, offR1, hb, wave, lane, ty, tx, false);
  __syncthreads();

  // ---- pool: 16 patches of 16x16 -> feat[img][j*16+i] ----
  {
    int patch = tid >> 4;
    int row   = tid & 15;
    int pr = patch >> 2, pc = patch & 3;
    const float* hp = s_h3 + ((pr * 16 + row) * TILE + pc * 16);
    float s = 0.0f;
#pragma unroll
    for (int k2 = 0; k2 < 16; ++k2) s += hp[k2];
    s_psum[tid] = s;
  }
  __syncthreads();
  if (tid < 16) {
    float t = 0.0f;
#pragma unroll
    for (int k2 = 0; k2 < 16; ++k2) t += s_psum[tid * 16 + k2];
    int gj = (ty >> 4) + (tid >> 2);
    int gi = (tx >> 4) + (tid & 3);
    feat[img * 256 + gj * 16 + gi] = t * (1.0f / 256.0f);
  }
}

// ---------------------------------------------------------------------------
// Kernel B: feat(256x256) @ wl1^T + bl1 -> e(256x64) @ wl2^T + bl2 -> softmax.
// Both GEMMs on WMMA f16 with f32 accumulate. wl2/bl1/bl2 arrive in LDS via
// the CDNA5 async global->LDS path (ASYNCcnt).
// ---------------------------------------------------------------------------
__global__ __launch_bounds__(NTHREADS)
void head_wmma_kernel(const float* __restrict__ feat,
                      const float* __restrict__ wl1, const float* __restrict__ bl1,
                      const float* __restrict__ wl2, const float* __restrict__ bl2,
                      float* __restrict__ out) {
  __shared__ __align__(32) _Float16 s_wB1[4 * 8 * 32 * 16]; // 32 KB
  __shared__ __align__(32) _Float16 s_e  [256 * 64];        // 32 KB
  __shared__ __align__(32) _Float16 s_wB2[2 * 32 * 16];     //  2 KB
  __shared__ float s_logits[256 * 8];                       //  8 KB
  __shared__ float s_c[512 + 64 + 8];                       // wl2 | bl1 | bl2

  const int tid  = threadIdx.x;
  const int wave = tid >> 5;
  const int lane = tid & 31;
  const int m    = lane & 15;
  const int kb   = (lane & 16) ? 8 : 0;      // A-operand K packing

  for (int i = tid; i < 512; i += NTHREADS) async_ld_f32(s_c + i,       wl2, i);
  for (int i = tid; i < 64;  i += NTHREADS) async_ld_f32(s_c + 512 + i, bl1, i);
  for (int i = tid; i < 8;   i += NTHREADS) async_ld_f32(s_c + 576 + i, bl2, i);

  for (int idx = tid; idx < 4 * 8 * 32 * 16; idx += NTHREADS) {
    int i      = idx & 15;
    int lane_b = (idx >> 4) & 31;
    int kc     = (idx >> 9) & 7;
    int nt     = idx >> 12;
    int n  = nt * 16 + (lane_b & 15);
    int kl = ((lane_b & 16) ? 8 : 0) + (i < 8 ? i : i + 8);
    int k  = kc * 32 + kl;
    s_wB1[idx] = (_Float16)wl1[n * 256 + k];
  }

  wait_asynccnt0();
  __syncthreads();

  for (int idx = tid; idx < 2 * 32 * 16; idx += NTHREADS) {
    int i      = idx & 15;
    int lane_b = (idx >> 4) & 31;
    int kc     = idx >> 9;
    int n  = lane_b & 15;
    int kl = ((lane_b & 16) ? 8 : 0) + (i < 8 ? i : i + 8);
    int k  = kc * 32 + kl;
    float v = (n < 8) ? s_c[n * 64 + k] : 0.0f;
    s_wB2[idx] = (_Float16)v;
  }
  __syncthreads();

  // GEMM1: 256x64x256, 16 M-tiles x 4 N-tiles, 8 tile-tasks per wave
  for (int task = wave * 8; task < wave * 8 + 8; ++task) {
    int mt = task >> 2, nt = task & 3;
    int row = mt * 16 + m;
    v8f acc = {};
#pragma unroll
    for (int kc = 0; kc < 8; ++kc) {
      const float* fp = feat + row * 256 + kc * 32;
      v16h a;
#pragma unroll
      for (int i = 0; i < 8; ++i) a[i]     = (_Float16)fp[kb + i];
#pragma unroll
      for (int i = 0; i < 8; ++i) a[8 + i] = (_Float16)fp[16 + kb + i];
      v16h b = *(const v16h*)(s_wB1 + (((nt * 8 + kc) * 32 + lane) << 4));
      acc = __builtin_amdgcn_wmma_f32_16x16x32_f16(false, a, false, b,
                                                   (short)0, acc, false, false);
    }
    int n    = nt * 16 + (lane & 15);
    float bi = s_c[512 + n];
    int mb   = (lane & 16) ? 8 : 0;
#pragma unroll
    for (int j = 0; j < 8; ++j)
      s_e[(mt * 16 + mb + j) * 64 + n] = (_Float16)(acc[j] + bi);
  }
  __syncthreads();

  // GEMM2: 256x8x64 (N padded to 16), 2 M-tiles per wave
  for (int task = wave * 2; task < wave * 2 + 2; ++task) {
    int row = task * 16 + m;
    v8f acc = {};
#pragma unroll
    for (int kc = 0; kc < 2; ++kc) {
      const _Float16* ep = s_e + row * 64 + kc * 32;
      union { v16h v; v8h h[2]; } a;
      a.h[0] = *(const v8h*)(ep + kb);
      a.h[1] = *(const v8h*)(ep + 16 + kb);
      v16h b = *(const v16h*)(s_wB2 + ((kc * 32 + lane) << 4));
      acc = __builtin_amdgcn_wmma_f32_16x16x32_f16(false, a.v, false, b,
                                                   (short)0, acc, false, false);
    }
    int n = lane & 15;
    if (n < 8) {
      float bi = s_c[576 + n];
      int mb   = (lane & 16) ? 8 : 0;
#pragma unroll
      for (int j = 0; j < 8; ++j)
        s_logits[(task * 16 + mb + j) * 8 + n] = acc[j] + bi;
    }
  }
  __syncthreads();

  // softmax over 8 logits, one row per thread
  {
    float l[8];
    float mx = -1e30f;
#pragma unroll
    for (int j = 0; j < 8; ++j) { l[j] = s_logits[tid * 8 + j]; mx = fmaxf(mx, l[j]); }
    float sum = 0.0f;
#pragma unroll
    for (int j = 0; j < 8; ++j) { l[j] = __expf(l[j] - mx); sum += l[j]; }
    float inv = 1.0f / sum;
#pragma unroll
    for (int j = 0; j < 8; ++j) out[tid * 8 + j] = l[j] * inv;
  }
}

// ---------------------------------------------------------------------------
extern "C" void kernel_launch(void* const* d_in, const int* in_sizes, int n_in,
                              void* d_out, int out_size, void* d_ws, size_t ws_size,
                              hipStream_t stream) {
  (void)in_sizes; (void)n_in; (void)out_size; (void)ws_size;
  const float* x   = (const float*)d_in[0];
  const float* w1  = (const float*)d_in[1];
  const float* b1  = (const float*)d_in[2];
  const float* w2  = (const float*)d_in[3];
  const float* b2  = (const float*)d_in[4];
  const float* w3  = (const float*)d_in[5];
  const float* b3  = (const float*)d_in[6];
  const float* wl1 = (const float*)d_in[7];
  const float* bl1 = (const float*)d_in[8];
  const float* wl2 = (const float*)d_in[9];
  const float* bl2 = (const float*)d_in[10];
  float* out  = (float*)d_out;
  float* feat = (float*)d_ws;               // 256*256 fp32 = 256 KB scratch

  dim3 gridA(16, 256);                      // 16 tiles x 256 images
  conv_pool_fused_kernel<<<gridA, NTHREADS, 0, stream>>>(x, w1, b1, w2, b2, w3, b3, feat);
  head_wmma_kernel<<<1, NTHREADS, 0, stream>>>(feat, wl1, bl1, wl2, bl2, out);
}
